// graphConvNet_3109556322840
// MI455X (gfx1250) — matrix operand order repaired and checked by reading, not compiled, not run on hardware
//
#include <hip/hip_runtime.h>

// ---------------------------------------------------------------------------
// GraphConvNet for MI455X (gfx1250, wave32, WMMA).
// Hot path: Y = diag(d) * A * (diag(d) * X) via v_wmma_f32_16x16x32_bf16.
// A (8192x8192 f32) is streamed and converted to bf16 in-register; activations
// are kept transposed in bf16 so the WMMA B operand is one contiguous 32B load
// per lane. K is split 4-way across blocks for occupancy; partials are summed
// in fixed order (deterministic, no float atomics).
// ---------------------------------------------------------------------------

typedef __attribute__((ext_vector_type(16))) __bf16 v16bf;
typedef __attribute__((ext_vector_type(8)))  float  v8f;
typedef __attribute__((ext_vector_type(4)))  float  v4f;

#define NV     8192
#define DIN    128
#define KSPLIT 4

// d[i] = rsqrt(D[i,i])
__global__ __launch_bounds__(256)
void k_diag_rsqrt(const float* __restrict__ D, float* __restrict__ d, int n) {
  int i = blockIdx.x * 256 + threadIdx.x;
  if (i < n) d[i] = rsqrtf(D[(size_t)i * n + i]);
}

// XsT0[c][j] = bf16(d[j] * H[j][c])   (transposed, column-scaled input)
__global__ __launch_bounds__(256)
void k_prep_h(const float* __restrict__ H, const float* __restrict__ d,
              __bf16* __restrict__ xsT, int n) {
  int j = blockIdx.x * 256 + threadIdx.x;
  int c = blockIdx.y;
  if (j < n) xsT[(size_t)c * n + j] = (__bf16)(d[j] * H[(size_t)j * DIN + c]);
}

// ---------------------------------------------------------------------------
// Ypart[kq][N x C] = diag(d) * A[:, krange] * Xs[krange, :]   (bf16 WMMA)
// Block: 4 waves, 64 rows; wave = 16-row strip x C/16 tiles; grid.y = KSPLIT.
// WMMA operand layouts used here (A/B: contiguous k, not ISA table order; C/D per ISA 7.12.2):
//   A 16x32 bf16: lane m (<16) holds row m, K 0..15 consecutive;
//                 lanes 16-31 hold K 16..31.
//   B 32x16 bf16: lane n (<16) holds column n, K 0..15 consecutive;
//                 lanes 16-31 hold K 16..31.
//   C/D f32: vgpr r -> M=r (lanes 0-15) / M=r+8 (lanes 16-31), N = lane%16.
// Inner loop: rolling 2-deep B-fragment pipeline + next-k A preload.
// ---------------------------------------------------------------------------
template <int C>
__global__ __launch_bounds__(128)
void k_gemm_an(const float* __restrict__ A, const float* __restrict__ d,
               const __bf16* __restrict__ xsT, float* __restrict__ Ypart,
               int n) {
  constexpr int T = C / 16;
  const int lane  = threadIdx.x & 31;
  const int wave  = threadIdx.x >> 5;
  const int mrow  = lane & 15;
  const int khalf = lane >> 4;
  const int row_base = blockIdx.x * 64 + wave * 16;
  const int kq   = blockIdx.y;
  const int kbeg = kq * (n / KSPLIT);
  const int kend = kbeg + n / KSPLIT;

  v8f acc[T];
#pragma unroll
  for (int t = 0; t < T; ++t)
    acc[t] = (v8f){0.f, 0.f, 0.f, 0.f, 0.f, 0.f, 0.f, 0.f};

  const float* arow = A + (size_t)(row_base + mrow) * n + khalf * 16;
  auto bptr = [&](int t, int k) {
    return (const v16bf*)(xsT + (size_t)(t * 16 + mrow) * n + k + khalf * 16);
  };

  // Prologue: A floats for kbeg, first two B fragments.
  v4f f0 = *(const v4f*)(arow + kbeg);
  v4f f1 = *(const v4f*)(arow + kbeg + 4);
  v4f f2 = *(const v4f*)(arow + kbeg + 8);
  v4f f3 = *(const v4f*)(arow + kbeg + 12);
  v16bf b[2];
  b[0] = *bptr(0, kbeg);
  b[1] = *bptr(1, kbeg);

  for (int k0 = kbeg; k0 < kend; k0 += 32) {
    // Convert current A chunk to the 16x32 bf16 fragment layout.
    v16bf a;
#pragma unroll
    for (int i = 0; i < 4; ++i) {
      a[i]      = (__bf16)f0[i];
      a[4 + i]  = (__bf16)f1[i];
      a[8 + i]  = (__bf16)f2[i];
      a[12 + i] = (__bf16)f3[i];
    }
    const int kn = (k0 + 32 < kend) ? k0 + 32 : kbeg;  // wrap: dummy reload
    // Stream A ahead of use (speculative prefetch of the HBM-bound stream).
    __builtin_prefetch(arow + k0 + 256, 0, 1);
    // Preload next k-step's A floats; they overlap the WMMA burst below.
    f0 = *(const v4f*)(arow + kn);
    f1 = *(const v4f*)(arow + kn + 4);
    f2 = *(const v4f*)(arow + kn + 8);
    f3 = *(const v4f*)(arow + kn + 12);
#pragma unroll
    for (int t = 0; t < T; ++t) {
      // Issue the load for tile t+2 (wrapping into next k-step), then WMMA.
      const int tn   = t + 2;
      const int tile = (tn < T) ? tn : tn - T;
      const int kl   = (tn < T) ? k0 : kn;
      v16bf bn = *bptr(tile, kl);
      acc[t] = __builtin_amdgcn_wmma_f32_16x16x32_bf16(
          false, a, false, b[t & 1], (short)0, acc[t], false, false);
      b[t & 1] = bn;
    }
  }

  float* yp = Ypart + (size_t)kq * n * C;
  const int rofs = khalf * 8;
#pragma unroll
  for (int t = 0; t < T; ++t) {
#pragma unroll
    for (int r = 0; r < 8; ++r) {
      int row = row_base + rofs + r;
      yp[(size_t)row * C + t * 16 + mrow] = acc[t][r] * d[row];
    }
  }
}

// ---------------------------------------------------------------------------
// Sum K-split partials (fixed order -> deterministic), Z = relu(Y @ W), then
// either XsT_next[cout][j] = bf16(d[j]*Z) or write X3 rows for the mean pass.
// Block: 16 rows x COUT threads.
// ---------------------------------------------------------------------------
template <int CIN, int COUT, bool LAST>
__global__ __launch_bounds__(1024)
void k_applyW(const float* __restrict__ Ypart, const float* __restrict__ W,
              const float* __restrict__ d, __bf16* __restrict__ xsT_next,
              float* __restrict__ x3out, int n) {
  __shared__ float Wl[CIN * COUT];
  __shared__ float Yl[16 * CIN];
  const int tx = threadIdx.x;            // 0..COUT-1
  const int ty = threadIdx.y;            // 0..15
  const int tid = ty * COUT + tx;
  const int nthr = COUT * 16;
  const int row0 = blockIdx.x * 16;
  for (int i = tid; i < CIN * COUT; i += nthr) Wl[i] = W[i];
  for (int i = tid; i < 16 * CIN; i += nthr) {
    size_t base = (size_t)(row0 + i / CIN) * CIN + (i % CIN);
    float s = 0.0f;
#pragma unroll
    for (int q = 0; q < KSPLIT; ++q) s += Ypart[(size_t)q * n * CIN + base];
    Yl[i] = s;
  }
  __syncthreads();
  float acc = 0.0f;
#pragma unroll 8
  for (int c = 0; c < CIN; ++c)
    acc = fmaf(Yl[ty * CIN + c], Wl[c * COUT + tx], acc);
  float z = fmaxf(acc, 0.0f);
  int row = row0 + ty;
  if (LAST) {
    x3out[(size_t)row * COUT + tx] = z;
  } else {
    xsT_next[(size_t)tx * n + row] = (__bf16)(d[row] * z);
  }
}

// Deterministic column mean of X3 [n x 16]: one block per column, fixed-order
// strided partial sums + LDS tree reduction.
__global__ __launch_bounds__(256)
void k_colmean(const float* __restrict__ X3, float* __restrict__ gout, int n) {
  __shared__ float red[256];
  const int c = blockIdx.x;  // 0..15
  float s = 0.0f;
  for (int i = threadIdx.x; i < n; i += 256) s += X3[(size_t)i * 16 + c];
  red[threadIdx.x] = s;
  __syncthreads();
  for (int w = 128; w > 0; w >>= 1) {
    if (threadIdx.x < w) red[threadIdx.x] += red[threadIdx.x + w];
    __syncthreads();
  }
  if (threadIdx.x == 0) gout[c] = red[0] / (float)n;
}

// ---------------------------------------------------------------------------
// Combine head: r[k] = sum_ij x_i cw1[i,j,k] y_j ; p = cw2 @ [x;y];
// z = relu(r+p+cbias); 16->8->4->1 MLP.
// ---------------------------------------------------------------------------
__global__ __launch_bounds__(32)
void k_combine(const float* __restrict__ gx, const float* __restrict__ gy,
               const float* __restrict__ cw1, const float* __restrict__ cw2,
               const float* __restrict__ cbias,
               const float* __restrict__ fw1, const float* __restrict__ fb1,
               const float* __restrict__ fw2, const float* __restrict__ fb2,
               const float* __restrict__ fw3, const float* __restrict__ fb3,
               float* __restrict__ out) {
  __shared__ float z[16];
  int k = threadIdx.x;
  if (k < 16) {
    float r = 0.0f;
    for (int i = 0; i < 16; ++i) {
      float xi = gx[i];
      for (int j = 0; j < 16; ++j)
        r = fmaf(xi * gy[j], cw1[(i * 16 + j) * 16 + k], r);
    }
    float p = 0.0f;
    for (int m = 0; m < 16; ++m) p = fmaf(cw2[k * 32 + m], gx[m], p);
    for (int m = 0; m < 16; ++m) p = fmaf(cw2[k * 32 + 16 + m], gy[m], p);
    z[k] = fmaxf(r + p + cbias[k], 0.0f);
  }
  __syncthreads();
  if (k == 0) {
    float z1[8], z2[4];
    for (int a = 0; a < 8; ++a) {
      float s = fb1[a];
      for (int q = 0; q < 16; ++q) s = fmaf(fw1[a * 16 + q], z[q], s);
      z1[a] = fmaxf(s, 0.0f);
    }
    for (int b = 0; b < 4; ++b) {
      float s = fb2[b];
      for (int a = 0; a < 8; ++a) s = fmaf(fw2[b * 8 + a], z1[a], s);
      z2[b] = fmaxf(s, 0.0f);
    }
    float s = fb3[0];
    for (int b = 0; b < 4; ++b) s = fmaf(fw3[b], z2[b], s);
    out[0] = fmaxf(s, 0.0f);
  }
}

static void run_graph(const float* A, const float* D, const float* H,
                      const float* W1, const float* W2, const float* W3,
                      float* dvec, __bf16* xsA, __bf16* xsB, float* Ypart,
                      float* x3buf, float* gout, hipStream_t stream) {
  const int n = NV;
  k_diag_rsqrt<<<n / 256, 256, 0, stream>>>(D, dvec, n);
  k_prep_h<<<dim3(n / 256, DIN), 256, 0, stream>>>(H, dvec, xsA, n);
  // layer 1: An @ (dH) -> Y[N,128]; relu(Y@W1) -> xsB (64 ch, scaled, T)
  k_gemm_an<128><<<dim3(n / 64, KSPLIT), 128, 0, stream>>>(A, dvec, xsA, Ypart, n);
  k_applyW<128, 64, false><<<n / 16, dim3(64, 16), 0, stream>>>(Ypart, W1, dvec, xsB, nullptr, n);
  // layer 2
  k_gemm_an<64><<<dim3(n / 64, KSPLIT), 128, 0, stream>>>(A, dvec, xsB, Ypart, n);
  k_applyW<64, 32, false><<<n / 16, dim3(32, 16), 0, stream>>>(Ypart, W2, dvec, xsA, nullptr, n);
  // layer 3 + deterministic column mean
  k_gemm_an<32><<<dim3(n / 64, KSPLIT), 128, 0, stream>>>(A, dvec, xsA, Ypart, n);
  k_applyW<32, 16, true><<<n / 16, dim3(16, 16), 0, stream>>>(Ypart, W3, dvec, nullptr, x3buf, n);
  k_colmean<<<16, 256, 0, stream>>>(x3buf, gout, n);
}

extern "C" void kernel_launch(void* const* d_in, const int* in_sizes, int n_in,
                              void* d_out, int out_size, void* d_ws, size_t ws_size,
                              hipStream_t stream) {
  const float* A1 = (const float*)d_in[0];
  const float* D1 = (const float*)d_in[1];
  const float* H1 = (const float*)d_in[2];
  const float* A2 = (const float*)d_in[3];
  const float* D2 = (const float*)d_in[4];
  const float* H2 = (const float*)d_in[5];
  const float* W1 = (const float*)d_in[6];
  const float* W2 = (const float*)d_in[7];
  const float* W3 = (const float*)d_in[8];
  const float* cw1 = (const float*)d_in[9];
  const float* cw2 = (const float*)d_in[10];
  const float* cbias = (const float*)d_in[11];
  const float* fw1 = (const float*)d_in[12];
  const float* fb1 = (const float*)d_in[13];
  const float* fw2 = (const float*)d_in[14];
  const float* fb2 = (const float*)d_in[15];
  const float* fw3 = (const float*)d_in[16];
  const float* fb3 = (const float*)d_in[17];

  char* ws = (char*)d_ws;
  size_t off = 0;
  auto alloc = [&](size_t bytes) -> void* {
    void* p = ws + off;
    off = (off + bytes + 255) & ~(size_t)255;
    return p;
  };
  float*  dvec  = (float*)alloc((size_t)NV * sizeof(float));
  __bf16* xsA   = (__bf16*)alloc((size_t)128 * NV * sizeof(__bf16));
  __bf16* xsB   = (__bf16*)alloc((size_t)128 * NV * sizeof(__bf16));
  float*  Ypart = (float*)alloc((size_t)KSPLIT * NV * 128 * sizeof(float));
  float*  x3buf = (float*)alloc((size_t)NV * 16 * sizeof(float));
  float*  g     = (float*)alloc(64 * sizeof(float));  // gx = g, gy = g+16

  run_graph(A1, D1, H1, W1, W2, W3, dvec, xsA, xsB, Ypart, x3buf, g, stream);
  run_graph(A2, D2, H2, W1, W2, W3, dvec, xsA, xsB, Ypart, x3buf, g + 16, stream);
  k_combine<<<1, 32, 0, stream>>>(g, g + 16, cw1, cw2, cbias,
                                  fw1, fb1, fw2, fb2, fw3, fb3, (float*)d_out);
}
